// LigRecDynamics_73031623901837
// MI455X (gfx1250) — matrix operand order repaired and checked by reading, not compiled, not run on hardware
//
#include <hip/hip_runtime.h>
#include <cstddef>
#include <cstdint>

// ---------------------------------------------------------------------------
// EGNN ligand-receptor dynamics forward pass for gfx1250 (MI455X).
// Heavy edge/node MLPs are batched into bf16 WMMA GEMMs (v_wmma_f32_16x16x32_bf16,
// f32 accumulate). Tiles are staged with GLOBAL_LOAD_ASYNC_TO_LDS_B128 into
// double-buffered LDS so the next K-step's loads overlap the current WMMAs.
// ---------------------------------------------------------------------------

typedef __attribute__((ext_vector_type(16))) __bf16 v16bf;
typedef __attribute__((ext_vector_type(8)))  __bf16 v8bf;
typedef __attribute__((ext_vector_type(8)))  float  v8f;
typedef __attribute__((ext_vector_type(4)))  int    v4i;

#if defined(__gfx1250__) && __has_builtin(__builtin_amdgcn_global_load_async_to_lds_b128)
#define EGNN_ASYNC_LDS 1
typedef __attribute__((address_space(1))) v4i as1_v4i;   // global
typedef __attribute__((address_space(3))) v4i as3_v4i;   // LDS
#else
#define EGNN_ASYNC_LDS 0
#endif

namespace {

constexpr int Bn  = 64;   // batch
constexpr int NLn = 64;   // ligand atoms
constexpr int NRn = 48;   // receptor atoms
constexpr int HFc = 257;  // HID+1
constexpr int PHc = 320;  // padded HF (multiple of 64, usable as K multiple of 32)
constexpr int KPE = 544;  // padded 515 / 514 (17*32)
constexpr int KLIG = 8, KREC = 6, NLAY = 4;
constexpr int RLL = Bn*NLn*KLIG;  // 32768 lig-lig edge rows
constexpr int RRL = Bn*NRn*KREC;  // 18432 rec-lig edge rows
constexpr int RND = Bn*NLn;       // 4096 node rows
constexpr int RRC = Bn*NRn;       // 3072 receptor rows

__device__ __forceinline__ float silu_f(float x){ return x / (1.f + __expf(-x)); }

// 16-byte memory -> LDS copy: async DMA on gfx1250, sync fallback elsewhere.
__device__ __forceinline__ void ld_lds16(const __bf16* g, __bf16* l)
{
#if EGNN_ASYNC_LDS
    __builtin_amdgcn_global_load_async_to_lds_b128(
        (as1_v4i*)(g), (as3_v4i*)(l), 0, 0);
#else
    *(v8bf*)l = *(const v8bf*)g;
#endif
}

__device__ __forceinline__ void wait_async0()
{
#if EGNN_ASYNC_LDS
#if __has_builtin(__builtin_amdgcn_s_wait_asynccnt)
    __builtin_amdgcn_s_wait_asynccnt(0);
#else
    asm volatile("s_wait_asynccnt 0x0" ::: "memory");
#endif
#endif
}

// ---------------- WMMA GEMM: C = act(A[MxK] * W[KxN] + bias) ----------------
// A: bf16 row-major, lda = padded K (multiple of 32).
// WT: bf16, N-major [Np x K] (pre-transposed weights), ldw = padded K.
// Cout: bf16 or f32, stride ldc. M % 128 == 0, N covered by grid.x*64.
// 256 threads = 8 waves; block tile 128x64; wave tile 16x64 (4 accumulators).
template<int OUT_BF16, int SILU>
__global__ void __launch_bounds__(256)
k_gemm(const __bf16* __restrict__ A, int lda,
       const __bf16* __restrict__ WT, int ldw,
       const float* __restrict__ bias,
       void* __restrict__ Cout, int ldc, int Ksteps)
{
    __shared__ __bf16 sA[2][128 * 32];
    __shared__ __bf16 sB[2][64 * 32];
    const int tid  = threadIdx.x;
    const int lane = tid & 31;
    const int wv   = tid >> 5;              // wave 0..7 -> M sub-tile
    const int m0   = blockIdx.y * 128;
    const int n0   = blockIdx.x * 64;
    const int lr   = tid >> 2;              // 0..63 staging row
    const int lc   = (tid & 3) * 8;         // 0,8,16,24 staging col

    // per-thread staging source pointers (advance by kb elements)
    const __bf16* gA0 = &A[(size_t)(m0 + lr)      * lda + lc];
    const __bf16* gA1 = &A[(size_t)(m0 + lr + 64) * lda + lc];
    const __bf16* gB  = &WT[(size_t)(n0 + lr)     * ldw + lc];

    v8f acc0 = {0,0,0,0,0,0,0,0};
    v8f acc1 = {0,0,0,0,0,0,0,0};
    v8f acc2 = {0,0,0,0,0,0,0,0};
    v8f acc3 = {0,0,0,0,0,0,0,0};

    const int arow = wv * 16 + (lane & 15);
    const int ac0  = (lane >> 4) * 16;      // A frag: K = ac0..ac0+15 contiguous
    const int bc0  = (lane >> 4) * 16;      // B frag: K = bc0..bc0+15 contiguous

    // prologue: stage tile 0 (asynchronously on gfx1250)
    ld_lds16(gA0, &sA[0][lr * 32 + lc]);
    ld_lds16(gA1, &sA[0][(lr + 64) * 32 + lc]);
    ld_lds16(gB,  &sB[0][lr * 32 + lc]);

    for (int ks = 0; ks < Ksteps; ++ks) {
        wait_async0();          // own tile-ks async loads landed in LDS
        __syncthreads();        // everyone's tile-ks staged; prev reads retired
        const int cur = ks & 1;
        if (ks + 1 < Ksteps) {  // overlap next tile's DMA with this tile's WMMAs
            const int nxt = cur ^ 1;
            const int kb  = (ks + 1) * 32;
            ld_lds16(gA0 + kb, &sA[nxt][lr * 32 + lc]);
            ld_lds16(gA1 + kb, &sA[nxt][(lr + 64) * 32 + lc]);
            ld_lds16(gB  + kb, &sB[nxt][lr * 32 + lc]);
        }

        const __bf16* a_ = &sA[cur][0];
        const __bf16* b_ = &sB[cur][0];
        union { v16bf v; v8bf h[2]; } au;  // 16-bit A layout: K {c0..c0+15}
        au.h[0] = *(const v8bf*)&a_[arow * 32 + ac0];
        au.h[1] = *(const v8bf*)&a_[arow * 32 + ac0 + 8];
        v16bf b0 = *(const v16bf*)&b_[((lane & 15))      * 32 + bc0];
        v16bf b1 = *(const v16bf*)&b_[(16 + (lane & 15)) * 32 + bc0];
        v16bf b2 = *(const v16bf*)&b_[(32 + (lane & 15)) * 32 + bc0];
        v16bf b3 = *(const v16bf*)&b_[(48 + (lane & 15)) * 32 + bc0];

        acc0 = __builtin_amdgcn_wmma_f32_16x16x32_bf16(false, au.v, false, b0, (short)0, acc0, false, false);
        acc1 = __builtin_amdgcn_wmma_f32_16x16x32_bf16(false, au.v, false, b1, (short)0, acc1, false, false);
        acc2 = __builtin_amdgcn_wmma_f32_16x16x32_bf16(false, au.v, false, b2, (short)0, acc2, false, false);
        acc3 = __builtin_amdgcn_wmma_f32_16x16x32_bf16(false, au.v, false, b3, (short)0, acc3, false, false);
    }

    const int mrow = m0 + wv * 16 + ((lane >> 4) << 3);  // C layout: M = r + (lane>=16)*8
    const int ncb  = n0 + (lane & 15);
    v8f acc[4] = {acc0, acc1, acc2, acc3};
#pragma unroll
    for (int j = 0; j < 4; ++j) {
        const int n = ncb + j * 16;
        const float bv = bias ? bias[n] : 0.f;
#pragma unroll
        for (int r = 0; r < 8; ++r) {
            float v = acc[j][r] + bv;
            if (SILU) v = silu_f(v);
            const size_t o = (size_t)(mrow + r) * ldc + n;
            if (OUT_BF16) ((__bf16*)Cout)[o] = (__bf16)v;
            else          ((float*)Cout)[o]  = v;
        }
    }
}

// ---------------- weight / bias / activation conversion ----------------
__global__ void k_conv_wt(const float* __restrict__ W, __bf16* __restrict__ WT,
                          int din, int dout, int Kp, int Np)
{
    int idx = blockIdx.x * 256 + threadIdx.x;
    if (idx >= Np * Kp) return;
    int n = idx / Kp, k = idx % Kp;
    float v = (n < dout && k < din) ? W[(size_t)k * dout + n] : 0.f;
    WT[idx] = (__bf16)v;
}

__global__ void k_conv_bias(const float* __restrict__ b, float* __restrict__ bp,
                            int dout, int Np)
{
    int i = blockIdx.x * 256 + threadIdx.x;
    if (i < Np) bp[i] = (i < dout) ? b[i] : 0.f;
}

__global__ void k_conv_act(const float* __restrict__ src, int ld, int ncols,
                           __bf16* __restrict__ dst, int Kp, int M)
{
    int idx = blockIdx.x * 256 + threadIdx.x;
    if (idx >= M * Kp) return;
    int m = idx / Kp, k = idx % Kp;
    dst[idx] = (__bf16)((k < ncols) ? src[(size_t)m * ld + k] : 0.f);
}

__global__ void k_copy(const float* __restrict__ s, float* __restrict__ d, int n)
{
    int i = blockIdx.x * 256 + threadIdx.x;
    if (i < n) d[i] = s[i];
}

// ---------------- kNN graph (built once from initial positions) ----------------
__global__ void k_build_graph(const float* __restrict__ lig_pos,
                              const float* __restrict__ rec_pos,
                              int* __restrict__ ll_src, int* __restrict__ rl_dst)
{
    const int b = blockIdx.x, t = threadIdx.x;
    __shared__ float lx[NLn * 3];
    if (t < NLn) {
        lx[t*3+0] = lig_pos[(b*NLn+t)*3+0];
        lx[t*3+1] = lig_pos[(b*NLn+t)*3+1];
        lx[t*3+2] = lig_pos[(b*NLn+t)*3+2];
    }
    __syncthreads();
    if (t < NLn) {
        unsigned long long used = 1ull << t;  // exclude self (matches +1e10 diag)
        const float x0 = lx[t*3+0], x1 = lx[t*3+1], x2 = lx[t*3+2];
        for (int k = 0; k < KLIG; ++k) {
            float best = 3.4e38f; int bj = 0;
            for (int j = 0; j < NLn; ++j) {
                if ((used >> j) & 1ull) continue;
                float d0 = lx[j*3+0]-x0, d1 = lx[j*3+1]-x1, d2 = lx[j*3+2]-x2;
                float d = d0*d0 + d1*d1 + d2*d2;
                if (d < best) { best = d; bj = j; }
            }
            ll_src[(b*NLn + t)*KLIG + k] = bj;
            used |= 1ull << bj;
        }
    }
    if (t < NRn) {
        const float r0 = rec_pos[(b*NRn+t)*3+0];
        const float r1 = rec_pos[(b*NRn+t)*3+1];
        const float r2 = rec_pos[(b*NRn+t)*3+2];
        unsigned long long used = 0;
        for (int k = 0; k < KREC; ++k) {
            float best = 3.4e38f; int bj = 0;
            for (int j = 0; j < NLn; ++j) {
                if ((used >> j) & 1ull) continue;
                float d0 = lx[j*3+0]-r0, d1 = lx[j*3+1]-r1, d2 = lx[j*3+2]-r2;
                float d = d0*d0 + d1*d1 + d2*d2;
                if (d < best) { best = d; bj = j; }
            }
            rl_dst[(b*NRn + t)*KREC + k] = bj;
            used |= 1ull << bj;
        }
    }
}

// append timestep channel (col 256), zero pad cols 257..319
__global__ void k_append_t(float* __restrict__ h, const float* __restrict__ ts, int natoms)
{
    int row = blockIdx.x, t = threadIdx.x;          // 64 threads
    int b = row / natoms;
    int c = 256 + t;
    if (c < PHc) h[(size_t)row * PHc + c] = (t == 0) ? ts[b] : 0.f;
}

// ---------------- edge feature gather (per layer; uses current coords) ----------------
__global__ void k_edge_ll(const float* __restrict__ h, const float* __restrict__ x,
                          const int* __restrict__ ll_src,
                          __bf16* __restrict__ F, float* __restrict__ xdn)
{
    const int e = blockIdx.x;                // 0..RLL-1
    const int b = e >> 9;                    // /512
    const int i = (e & 511) >> 3;            // dst atom
    const int s = ll_src[e];
    const float* xs = &x[(size_t)(b*NLn + s) * 3];
    const float* xd = &x[(size_t)(b*NLn + i) * 3];
    const float dx0 = xs[0]-xd[0], dx1 = xs[1]-xd[1], dx2 = xs[2]-xd[2];
    const float d = sqrtf(dx0*dx0 + dx1*dx1 + dx2*dx2);
    const float* hs = &h[(size_t)(b*NLn + s) * PHc];
    const float* hd = &h[(size_t)(b*NLn + i) * PHc];
    __bf16* Fr = &F[(size_t)e * KPE];
    for (int c = threadIdx.x; c < KPE; c += 64) {
        float v;
        if      (c < HFc)      v = hs[c];
        else if (c < 2*HFc)    v = hd[c - HFc];
        else if (c == 2*HFc)   v = d;
        else                   v = 0.f;
        Fr[c] = (__bf16)v;
    }
    if (threadIdx.x < 3) {
        float dc = threadIdx.x == 0 ? dx0 : (threadIdx.x == 1 ? dx1 : dx2);
        xdn[(size_t)e * 3 + threadIdx.x] = dc / (d + 1e-9f);
    }
}

__global__ void k_edge_rl(const float* __restrict__ hr, const float* __restrict__ h,
                          const float* __restrict__ x, const float* __restrict__ rec_pos,
                          const int* __restrict__ rl_dst,
                          __bf16* __restrict__ F, float* __restrict__ xdn)
{
    const int e = blockIdx.x;                // 0..RRL-1
    const int b = e / (NRn * KREC);
    const int el = e % (NRn * KREC);
    const int r = el / KREC;                 // src receptor atom
    const int i = rl_dst[e];                 // dst ligand atom
    const float* xs = &rec_pos[(size_t)(b*NRn + r) * 3];
    const float* xd = &x[(size_t)(b*NLn + i) * 3];
    const float dx0 = xs[0]-xd[0], dx1 = xs[1]-xd[1], dx2 = xs[2]-xd[2];
    const float d = sqrtf(dx0*dx0 + dx1*dx1 + dx2*dx2);
    const float* hs = &hr[(size_t)(b*NRn + r) * PHc];
    const float* hd = &h[(size_t)(b*NLn + i) * PHc];
    __bf16* Fr = &F[(size_t)e * KPE];
    for (int c = threadIdx.x; c < KPE; c += 64) {
        float v;
        if      (c < HFc)      v = hs[c];
        else if (c < 2*HFc)    v = hd[c - HFc];
        else if (c == 2*HFc)   v = d;
        else                   v = 0.f;
        Fr[c] = (__bf16)v;
    }
    if (threadIdx.x < 3) {
        float dc = threadIdx.x == 0 ? dx0 : (threadIdx.x == 1 ? dx1 : dx2);
        xdn[(size_t)e * 3 + threadIdx.x] = dc / (d + 1e-9f);
    }
}

// final coord projection (257 -> 1, no bias) and scaling by unit vector
__global__ void k_coord_scalar(const float* __restrict__ C2, const float* __restrict__ w2,
                               const float* __restrict__ xdn, float* __restrict__ mx, int nrows)
{
    int r = blockIdx.x * 256 + threadIdx.x;
    if (r >= nrows) return;
    const float* c = &C2[(size_t)r * PHc];
    float s = 0.f;
    for (int d = 0; d < HFc; ++d) s += c[d] * w2[d];
    mx[(size_t)r*3+0] = s * xdn[(size_t)r*3+0];
    mx[(size_t)r*3+1] = s * xdn[(size_t)r*3+1];
    mx[(size_t)r*3+2] = s * xdn[(size_t)r*3+2];
}

// segment-sum of edge messages into nodes; emit node-MLP input rows (bf16)
__global__ void k_aggregate(const float* __restrict__ Mh_ll, const float* __restrict__ Mh_rl,
                            const float* __restrict__ mx_ll, const float* __restrict__ mx_rl,
                            const int* __restrict__ rl_dst, const float* __restrict__ h,
                            __bf16* __restrict__ F_nd, float* __restrict__ x_neigh)
{
    const int row = blockIdx.x;              // b*64 + i
    const int b = row >> 6, i = row & 63;
    __shared__ int s_rd[NRn * KREC];         // 288
    const int t = threadIdx.x;               // 320 threads
    if (t < NRn * KREC) s_rd[t] = rl_dst[b * NRn * KREC + t];
    __syncthreads();
    if (t < HFc) {
        float acc = 0.f;
        const float* bl = &Mh_ll[(size_t)(b*512 + i*8) * PHc + t];
        for (int k = 0; k < KLIG; ++k) acc += bl[(size_t)k * PHc];
        for (int e = 0; e < NRn * KREC; ++e)
            if (s_rd[e] == i) acc += Mh_rl[(size_t)(b*288 + e) * PHc + t];
        __bf16* Fr = &F_nd[(size_t)row * KPE];
        Fr[t]       = (__bf16)h[(size_t)row * PHc + t];
        Fr[HFc + t] = (__bf16)acc;
    } else if (t < HFc + 3) {
        const int c = t - HFc;
        float acc = 0.f;
        for (int k = 0; k < KLIG; ++k) acc += mx_ll[(size_t)(b*512 + i*8 + k)*3 + c];
        for (int e = 0; e < NRn * KREC; ++e)
            if (s_rd[e] == i) acc += mx_rl[(size_t)(b*288 + e)*3 + c];
        x_neigh[(size_t)row*3 + c] = acc;
    } else if (t < HFc + 3 + (KPE - 2*HFc)) {    // zero pad cols 514..543
        F_nd[(size_t)row * KPE + 2*HFc + (t - (HFc + 3))] = (__bf16)0.f;
    }
}

__global__ void k_residual(float* __restrict__ h, const float* __restrict__ nout,
                           float* __restrict__ x, const float* __restrict__ xn)
{
    const int row = blockIdx.x, t = threadIdx.x;     // 320 threads
    if (t < HFc) h[(size_t)row * PHc + t] += nout[(size_t)row * PHc + t];
    else if (t < HFc + 3) {
        const int c = t - HFc;
        x[(size_t)row*3 + c] += xn[(size_t)row*3 + c];
    }
}

__global__ void k_output(const float* __restrict__ dec2, const float* __restrict__ x,
                         const float* __restrict__ lig_pos, float* __restrict__ out)
{
    const int row = blockIdx.x, t = threadIdx.x;     // 64 threads
    if (t < 32) out[(size_t)row * 32 + t] = dec2[(size_t)row * 64 + t];
    else if (t < 35) {
        const int c = t - 32;
        out[(size_t)Bn*NLn*32 + (size_t)row*3 + c] = x[(size_t)row*3+c] - lig_pos[(size_t)row*3+c];
    }
}

} // namespace

// ---------------------------------------------------------------------------
// host orchestration
// ---------------------------------------------------------------------------
extern "C" void kernel_launch(void* const* d_in, const int* in_sizes, int n_in,
                              void* d_out, int out_size, void* d_ws, size_t ws_size,
                              hipStream_t stream)
{
    const float* lig_pos  = (const float*)d_in[0];
    const float* lig_feat = (const float*)d_in[1];
    const float* rec_pos  = (const float*)d_in[2];
    const float* rec_feat = (const float*)d_in[3];
    const float* tsp      = (const float*)d_in[4];
    // param leaves (jax pytree order): layers[0..3]{ll_coord(5),ll_edge(4),node(4),
    // rl_coord(5),rl_edge(4)}, lig_dec(4), lig_enc(4), rec_enc(4)
    auto P = [&](int i) -> const float* { return (const float*)d_in[5 + i]; };

    char* wsb = (char*)d_ws; size_t off = 0;
    auto alloc = [&](size_t b) -> void* {
        void* p = wsb + off; off = (off + b + 255) & ~(size_t)255; return p;
    };

    // activations / graph / scratch
    float*  hl     = (float*)alloc((size_t)RND * PHc * 4);
    float*  hr     = (float*)alloc((size_t)RRC * PHc * 4);
    float*  xc     = (float*)alloc((size_t)RND * 3 * 4);
    float*  xn     = (float*)alloc((size_t)RND * 3 * 4);
    int*    ll_src = (int*)alloc((size_t)RLL * 4);
    int*    rl_dst = (int*)alloc((size_t)RRL * 4);
    __bf16* F_ll   = (__bf16*)alloc((size_t)RLL * KPE * 2);
    __bf16* F_rl   = (__bf16*)alloc((size_t)RRL * KPE * 2);
    __bf16* H1     = (__bf16*)alloc((size_t)RLL * PHc * 2);
    float*  C2     = (float*)alloc((size_t)RLL * PHc * 4);
    float*  Mh_ll  = (float*)alloc((size_t)RLL * PHc * 4);
    float*  Mh_rl  = (float*)alloc((size_t)RRL * PHc * 4);
    float*  xdn_ll = (float*)alloc((size_t)RLL * 3 * 4);
    float*  xdn_rl = (float*)alloc((size_t)RRL * 3 * 4);
    float*  mx_ll  = (float*)alloc((size_t)RLL * 3 * 4);
    float*  mx_rl  = (float*)alloc((size_t)RRL * 3 * 4);
    __bf16* F_nd   = (__bf16*)alloc((size_t)RND * KPE * 2);
    __bf16* Hn     = (__bf16*)alloc((size_t)RND * PHc * 2);
    float*  nout   = (float*)alloc((size_t)RND * PHc * 4);
    __bf16* Aenc   = (__bf16*)alloc((size_t)RND * 256 * 2);
    __bf16* Henc   = (__bf16*)alloc((size_t)RRC * 256 * 2);
    float*  dec2   = (float*)alloc((size_t)RND * 64 * 4);

    // per-layer converted weights (N-major bf16) + padded biases
    struct LW {
        __bf16 *llc0,*llc1,*lle0,*lle1,*rlc0,*rlc1,*rle0,*rle1,*n0,*n1;
        float *bllc0,*bllc1,*blle0,*blle1,*brlc0,*brlc1,*brle0,*brle1,*bn0,*bn1;
        const float *w2ll,*w2rl;
    } L[NLAY];
    for (int l = 0; l < NLAY; ++l) {
        L[l].llc0 = (__bf16*)alloc((size_t)PHc*KPE*2); L[l].llc1 = (__bf16*)alloc((size_t)PHc*PHc*2);
        L[l].lle0 = (__bf16*)alloc((size_t)PHc*KPE*2); L[l].lle1 = (__bf16*)alloc((size_t)PHc*PHc*2);
        L[l].rlc0 = (__bf16*)alloc((size_t)PHc*KPE*2); L[l].rlc1 = (__bf16*)alloc((size_t)PHc*PHc*2);
        L[l].rle0 = (__bf16*)alloc((size_t)PHc*KPE*2); L[l].rle1 = (__bf16*)alloc((size_t)PHc*PHc*2);
        L[l].n0   = (__bf16*)alloc((size_t)PHc*KPE*2); L[l].n1   = (__bf16*)alloc((size_t)PHc*PHc*2);
        L[l].bllc0=(float*)alloc(PHc*4); L[l].bllc1=(float*)alloc(PHc*4);
        L[l].blle0=(float*)alloc(PHc*4); L[l].blle1=(float*)alloc(PHc*4);
        L[l].brlc0=(float*)alloc(PHc*4); L[l].brlc1=(float*)alloc(PHc*4);
        L[l].brle0=(float*)alloc(PHc*4); L[l].brle1=(float*)alloc(PHc*4);
        L[l].bn0  =(float*)alloc(PHc*4); L[l].bn1  =(float*)alloc(PHc*4);
    }
    __bf16* Wlig0=(__bf16*)alloc((size_t)64*32*2);   float* blig0=(float*)alloc(64*4);
    __bf16* Wlig1=(__bf16*)alloc((size_t)256*64*2);  float* blig1=(float*)alloc(256*4);
    __bf16* Wrec0=(__bf16*)alloc((size_t)256*128*2); float* brec0=(float*)alloc(256*4);
    __bf16* Wrec1=(__bf16*)alloc((size_t)256*256*2); float* brec1=(float*)alloc(256*4);
    __bf16* Wdec0=(__bf16*)alloc((size_t)64*256*2);  float* bdec0=(float*)alloc(64*4);
    __bf16* Wdec1=(__bf16*)alloc((size_t)64*64*2);   float* bdec1=(float*)alloc(64*4);
    (void)ws_size; (void)in_sizes; (void)n_in; (void)out_size;

    auto convW = [&](const float* W, __bf16* WT, int din, int dout, int Kp, int Np) {
        int n = Np * Kp;
        k_conv_wt<<<(n + 255)/256, 256, 0, stream>>>(W, WT, din, dout, Kp, Np);
    };
    auto convB = [&](const float* b, float* bp, int dout, int Np) {
        k_conv_bias<<<(Np + 255)/256, 256, 0, stream>>>(b, bp, dout, Np);
    };
    auto gemm = [&](bool obf, bool silu, const __bf16* A, int lda, const __bf16* WT, int ldw,
                    const float* bias, void* C, int ldc, int M, int Np, int Ks) {
        dim3 g(Np/64, M/128), blk(256);
        if (obf) { if (silu) k_gemm<1,1><<<g,blk,0,stream>>>(A,lda,WT,ldw,bias,C,ldc,Ks);
                   else      k_gemm<1,0><<<g,blk,0,stream>>>(A,lda,WT,ldw,bias,C,ldc,Ks); }
        else     { if (silu) k_gemm<0,1><<<g,blk,0,stream>>>(A,lda,WT,ldw,bias,C,ldc,Ks);
                   else      k_gemm<0,0><<<g,blk,0,stream>>>(A,lda,WT,ldw,bias,C,ldc,Ks); }
    };

    // ---- convert all weights once per launch (deterministic) ----
    for (int l = 0; l < NLAY; ++l) {
        const int pb = 22 * l;
        convW(P(pb+0),  L[l].llc0, 515, 257, KPE, PHc); convB(P(pb+1),  L[l].bllc0, 257, PHc);
        convW(P(pb+2),  L[l].llc1, 257, 257, PHc, PHc); convB(P(pb+3),  L[l].bllc1, 257, PHc);
        L[l].w2ll = P(pb+4);
        convW(P(pb+5),  L[l].lle0, 515, 257, KPE, PHc); convB(P(pb+6),  L[l].blle0, 257, PHc);
        convW(P(pb+7),  L[l].lle1, 257, 257, PHc, PHc); convB(P(pb+8),  L[l].blle1, 257, PHc);
        convW(P(pb+9),  L[l].n0,   514, 257, KPE, PHc); convB(P(pb+10), L[l].bn0,   257, PHc);
        convW(P(pb+11), L[l].n1,   257, 257, PHc, PHc); convB(P(pb+12), L[l].bn1,   257, PHc);
        convW(P(pb+13), L[l].rlc0, 515, 257, KPE, PHc); convB(P(pb+14), L[l].brlc0, 257, PHc);
        convW(P(pb+15), L[l].rlc1, 257, 257, PHc, PHc); convB(P(pb+16), L[l].brlc1, 257, PHc);
        L[l].w2rl = P(pb+17);
        convW(P(pb+18), L[l].rle0, 515, 257, KPE, PHc); convB(P(pb+19), L[l].brle0, 257, PHc);
        convW(P(pb+20), L[l].rle1, 257, 257, PHc, PHc); convB(P(pb+21), L[l].brle1, 257, PHc);
    }
    convW(P(88), Wdec0, 256, 64, 256, 64);  convB(P(89), bdec0, 64, 64);
    convW(P(90), Wdec1, 64, 32, 64, 64);    convB(P(91), bdec1, 32, 64);
    convW(P(92), Wlig0, 32, 64, 32, 64);    convB(P(93), blig0, 64, 64);
    convW(P(94), Wlig1, 64, 256, 64, 256);  convB(P(95), blig1, 256, 256);
    convW(P(96), Wrec0, 128, 256, 128, 256);convB(P(97), brec0, 256, 256);
    convW(P(98), Wrec1, 256, 256, 256, 256);convB(P(99), brec1, 256, 256);

    // ---- init coords + kNN graphs ----
    k_copy<<<(RND*3 + 255)/256, 256, 0, stream>>>(lig_pos, xc, RND*3);
    k_build_graph<<<Bn, 64, 0, stream>>>(lig_pos, rec_pos, ll_src, rl_dst);

    // ---- encoders ----
    k_conv_act<<<(RND*32 + 255)/256, 256, 0, stream>>>(lig_feat, 32, 32, Aenc, 32, RND);
    gemm(true,  true,  Aenc, 32,  Wlig0, 32,  blig0, Henc, 64,  RND, 64,  1);
    gemm(false, false, Henc, 64,  Wlig1, 64,  blig1, hl,   PHc, RND, 256, 2);
    k_append_t<<<RND, 64, 0, stream>>>(hl, tsp, NLn);

    k_conv_act<<<(RRC*128 + 255)/256, 256, 0, stream>>>(rec_feat, 128, 128, Aenc, 128, RRC);
    gemm(true,  true,  Aenc, 128, Wrec0, 128, brec0, Henc, 256, RRC, 256, 4);
    gemm(false, false, Henc, 256, Wrec1, 256, brec1, hr,   PHc, RRC, 256, 8);
    k_append_t<<<RRC, 64, 0, stream>>>(hr, tsp, NRn);

    // ---- message-passing layers ----
    for (int l = 0; l < NLAY; ++l) {
        k_edge_ll<<<RLL, 64, 0, stream>>>(hl, xc, ll_src, F_ll, xdn_ll);
        k_edge_rl<<<RRL, 64, 0, stream>>>(hr, hl, xc, rec_pos, rl_dst, F_rl, xdn_rl);

        // ll coord path: F -> H1 -> C2 -> mx_ll
        gemm(true,  true, F_ll, KPE, L[l].llc0, KPE, L[l].bllc0, H1, PHc, RLL, PHc, 17);
        gemm(false, true, H1,   PHc, L[l].llc1, PHc, L[l].bllc1, C2, PHc, RLL, PHc, 10);
        k_coord_scalar<<<(RLL + 255)/256, 256, 0, stream>>>(C2, L[l].w2ll, xdn_ll, mx_ll, RLL);
        // ll edge path: F -> H1 -> Mh_ll
        gemm(true,  true, F_ll, KPE, L[l].lle0, KPE, L[l].blle0, H1,    PHc, RLL, PHc, 17);
        gemm(false, true, H1,   PHc, L[l].lle1, PHc, L[l].blle1, Mh_ll, PHc, RLL, PHc, 10);

        // rl coord path
        gemm(true,  true, F_rl, KPE, L[l].rlc0, KPE, L[l].brlc0, H1, PHc, RRL, PHc, 17);
        gemm(false, true, H1,   PHc, L[l].rlc1, PHc, L[l].brlc1, C2, PHc, RRL, PHc, 10);
        k_coord_scalar<<<(RRL + 255)/256, 256, 0, stream>>>(C2, L[l].w2rl, xdn_rl, mx_rl, RRL);
        // rl edge path
        gemm(true,  true, F_rl, KPE, L[l].rle0, KPE, L[l].brle0, H1,    PHc, RRL, PHc, 17);
        gemm(false, true, H1,   PHc, L[l].rle1, PHc, L[l].brle1, Mh_rl, PHc, RRL, PHc, 10);

        // aggregate + node MLP + residual
        k_aggregate<<<RND, 320, 0, stream>>>(Mh_ll, Mh_rl, mx_ll, mx_rl, rl_dst, hl, F_nd, xn);
        gemm(true,  true,  F_nd, KPE, L[l].n0, KPE, L[l].bn0, Hn,   PHc, RND, PHc, 17);
        gemm(false, false, Hn,   PHc, L[l].n1, PHc, L[l].bn1, nout, PHc, RND, PHc, 10);
        k_residual<<<RND, 320, 0, stream>>>(hl, nout, xc, xn);
    }

    // ---- decoder + outputs ----
    k_conv_act<<<(RND*256 + 255)/256, 256, 0, stream>>>(hl, PHc, 256, Aenc, 256, RND);
    gemm(true,  true,  Aenc, 256, Wdec0, 256, bdec0, Henc, 64, RND, 64, 8);
    gemm(false, false, Henc, 64,  Wdec1, 64,  bdec1, dec2, 64, RND, 64, 2);
    k_output<<<RND, 64, 0, stream>>>(dec2, xc, lig_pos, (float*)d_out);
}
